// ASTGCN_block_9036611191564
// MI455X (gfx1250) — matrix operand order repaired and checked by reading, not compiled, not run on hardware
//
#include <hip/hip_runtime.h>
#include <hip/hip_bf16.h>
#include <math.h>

// Shapes (fixed by the reference): B=32, N=1024, C=64, T=24, K=3, Fc=Ft=64
#define BB 32
#define NN 1024
#define CC 64
#define TT 24
#define CT 1536   // C*T
#define FC 64

typedef __attribute__((ext_vector_type(16))) __bf16 bf16x16;
typedef __attribute__((ext_vector_type(8)))  __bf16 bf16x8;
typedef __attribute__((ext_vector_type(8)))  float  f32x8;

#define LDK 40   // LDS row stride (bf16 elems): 80B rows, 16B-aligned chunks, low bank conflict

__device__ __forceinline__ float sigmoidf_(float v) { return 1.f / (1.f + __expf(-v)); }

// Load one 16x32 (A) or 32x16 (B) bf16 WMMA fragment from an LDS tile stored
// as [row][k] with row stride LDK. off0/off1 select the two 8-elem (16B) chunks
// this lane needs per the CDNA5 16-bit A/B VGPR layouts (CDNA5 ISA 7.12.2).
__device__ __forceinline__ bf16x16 ldsfrag(const __bf16* row, int off0, int off1) {
  bf16x8 lo = *(const bf16x8*)(row + off0);
  bf16x8 hi = *(const bf16x8*)(row + off1);
  return __builtin_shufflevector(lo, hi, 0,1,2,3,4,5,6,7,8,9,10,11,12,13,14,15);
}

// ---------------------------------------------------------------------------
// k1: a_tc[b,t,c] = sum_n U1[n] x[b,n,c,t]
//     r_ct[b,c,t] = sum_n U2[c,n] * (sum_c' U3[c'] x[b,n,c',t])
// Grid (16 nchunks, B), 256 thr. Atomic accumulation into pre-zeroed buffers.
// ---------------------------------------------------------------------------
__global__ void k1_temporal_pre(const float* __restrict__ x, const float* __restrict__ U1,
                                const float* __restrict__ U2, const float* __restrict__ U3,
                                float* __restrict__ a_tc, float* __restrict__ r_ct) {
  const int b = blockIdx.y, chunk = blockIdx.x, tid = threadIdx.x;
  __shared__ float sh[CT];
  __shared__ float shr[TT];
  float accA[6] = {0,0,0,0,0,0}, accR[6] = {0,0,0,0,0,0};
  int cc[6], tt[6];
  for (int s = 0; s < 6; ++s) { int e = s*256 + tid; cc[s] = e/TT; tt[s] = e%TT; }
  const float* xb = x + (size_t)b*NN*CT;
  for (int nl = 0; nl < 64; ++nl) {
    const int n = chunk*64 + nl;
    const float u1 = U1[n];
    float xv[6];
    for (int s = 0; s < 6; ++s) {
      int e = s*256 + tid;
      xv[s] = xb[(size_t)n*CT + e];
      accA[s] += u1 * xv[s];
      sh[e] = U3[cc[s]] * xv[s];
    }
    __syncthreads();
    if (tid < TT) {
      float r = 0.f;
      for (int c = 0; c < CC; ++c) r += sh[c*TT + tid];
      shr[tid] = r;
    }
    __syncthreads();
    for (int s = 0; s < 6; ++s) accR[s] += U2[cc[s]*NN + n] * shr[tt[s]];
    __syncthreads();
  }
  for (int s = 0; s < 6; ++s) {
    atomicAdd(&a_tc[b*CT + tt[s]*CC + cc[s]], accA[s]);
    atomicAdd(&r_ct[b*CT + cc[s]*TT + tt[s]], accR[s]);
  }
}

// ---------------------------------------------------------------------------
// k2: E[b] = softmax_axis1( sigmoid(a_tc@r_ct + b_e) @ V_e^T ); ew[b] = E@W1
// Grid B, 576 threads (one per (i,j) of 24x24).
// ---------------------------------------------------------------------------
__global__ void k2_E(const float* __restrict__ a_tc, const float* __restrict__ r_ct,
                     const float* __restrict__ b_e, const float* __restrict__ V_e,
                     const float* __restrict__ W1, float* __restrict__ E,
                     float* __restrict__ ew) {
  const int b = blockIdx.x, tid = threadIdx.x;
  __shared__ float a_l[CT], r_l[CT], s0[576], s1[576], colm[TT], cols[TT];
  for (int idx = tid; idx < CT; idx += 576) { a_l[idx] = a_tc[b*CT+idx]; r_l[idx] = r_ct[b*CT+idx]; }
  __syncthreads();
  const int i = tid / TT, j = tid % TT;
  {
    float p = 0.f;
    for (int c = 0; c < CC; ++c) p += a_l[i*CC+c] * r_l[c*TT+j];
    s0[tid] = sigmoidf_(p + b_e[i*TT+j]);
  }
  __syncthreads();
  {
    float ep = 0.f;
    for (int kk = 0; kk < TT; ++kk) ep += s0[i*TT+kk] * V_e[j*TT+kk];
    s1[tid] = ep;
  }
  __syncthreads();
  if (tid < TT) {
    float m = -1e30f;
    for (int ii = 0; ii < TT; ++ii) m = fmaxf(m, s1[ii*TT+tid]);
    float ss = 0.f;
    for (int ii = 0; ii < TT; ++ii) ss += __expf(s1[ii*TT+tid] - m);
    colm[tid] = m; cols[tid] = 1.f/ss;
  }
  __syncthreads();
  const float Ev = __expf(s1[tid]-colm[j]) * cols[j];
  E[b*576+tid] = Ev;
  s0[tid] = Ev;
  __syncthreads();
  if (tid < TT) {
    float e2 = 0.f;
    for (int t = 0; t < TT; ++t) e2 += s0[tid*TT+t] * W1[t];
    ew[b*TT+tid] = e2;
  }
}

// ---------------------------------------------------------------------------
// k3: fused x_TAt reductions (never materializes x_TAt):
//   lhs_s[b,n,t'] = sum_c (sum_u x[b,n,c,u] ew[b,u]) W2[c,t']
//   rhs_s[b,t,n]  = sum_u (sum_c W3[c] x[b,n,c,u]) E[b,u,t]
// Grid (N/8, B), 256 thr, 8 n per block staged in LDS.
// ---------------------------------------------------------------------------
__global__ void k3_spatial_pre(const float* __restrict__ x, const float* __restrict__ E,
                               const float* __restrict__ ew, const float* __restrict__ W2,
                               const float* __restrict__ W3, float* __restrict__ lhs_s,
                               float* __restrict__ rhs_s) {
  const int b = blockIdx.y, n0 = blockIdx.x*8, tid = threadIdx.x;
  __shared__ float xs[8*CT];
  __shared__ float Esh[576], ewsh[TT], W2sh[CT], W3sh[CC];
  __shared__ float lhsp[4][CC], xwsh[4][TT];
  const float* xb = x + ((size_t)b*NN + n0)*CT;
  for (int idx = tid; idx < 8*CT; idx += 256) xs[idx] = xb[idx];
  for (int idx = tid; idx < 576; idx += 256) Esh[idx] = E[b*576+idx];
  for (int idx = tid; idx < CT; idx += 256) W2sh[idx] = W2[idx];
  if (tid < CC) W3sh[tid] = W3[tid];
  if (tid < TT) ewsh[tid] = ew[b*TT+tid];
  __syncthreads();
  const int g = tid >> 6, l = tid & 63;
  for (int pass = 0; pass < 2; ++pass) {
    const int nl = pass*4 + g;
    const float* xn = &xs[nl*CT];
    float lp = 0.f;
    for (int u = 0; u < TT; ++u) lp += xn[l*TT+u] * ewsh[u];
    lhsp[g][l] = lp;
    if (l < TT) {
      float xw = 0.f;
      for (int c = 0; c < CC; ++c) xw += W3sh[c] * xn[c*TT+l];
      xwsh[g][l] = xw;
    }
    __syncthreads();
    if (l < TT) {
      float lo = 0.f, ro = 0.f;
      for (int c = 0; c < CC; ++c) lo += lhsp[g][c] * W2sh[c*TT+l];
      for (int u = 0; u < TT; ++u) ro += xwsh[g][u] * Esh[u*TT+l];
      lhs_s[((size_t)b*NN + n0+nl)*TT + l] = lo;
      rhs_s[((size_t)b*TT + l)*NN + n0+nl] = ro;
    }
    __syncthreads();
  }
}

// ---------------------------------------------------------------------------
// k4: prod_s[b,i,j] = sum_t lhs_s[b,i,t] rhs_s[b,t,j]. K=24 -> VALU tiles,
// memory-bound on the 134 MB write anyway. Grid (16 j, 16 i, B), 64x64 tiles.
// ---------------------------------------------------------------------------
__global__ void k4_prods(const float* __restrict__ lhs_s, const float* __restrict__ rhs_s,
                         float* __restrict__ prod_s) {
  const int b = blockIdx.z, i0 = blockIdx.y*64, j0 = blockIdx.x*64, tid = threadIdx.x;
  __shared__ float L[64*TT], R[TT*64];
  for (int idx = tid; idx < CT; idx += 256) L[idx] = lhs_s[((size_t)b*NN + i0)*TT + idx];
  for (int idx = tid; idx < CT; idx += 256) {
    int t = idx >> 6, j = idx & 63;
    R[idx] = rhs_s[((size_t)b*TT + t)*NN + j0 + j];
  }
  __syncthreads();
  const int ti = tid >> 4, tj = tid & 15;
  for (int a = 0; a < 4; ++a)
    for (int bq = 0; bq < 4; ++bq) {
      const int il = a*16+ti, jl = bq*16+tj;
      float acc = 0.f;
      for (int t = 0; t < TT; ++t) acc += L[il*TT+t] * R[t*64+jl];
      prod_s[((size_t)b*NN + i0+il)*NN + j0+jl] = acc;
    }
}

// ---------------------------------------------------------------------------
// k5a: S_pre[b,i,j] = sum_m sigmoid(prod_s+b_s)[i,m] * V_s[j,m]
// WMMA bf16 tiled GEMM: 128x128 block tile, 8 waves (4i x 2j), K-step 32.
// Sigmoid fused into the A-tile load; everything accumulated in f32.
// ---------------------------------------------------------------------------
__global__ void k5a_spre(const float* __restrict__ prod_s, const float* __restrict__ b_s,
                         const float* __restrict__ V_s, float* __restrict__ S) {
  const int b = blockIdx.z, i0 = blockIdx.y*128, j0 = blockIdx.x*128, tid = threadIdx.x;
  __shared__ alignas(16) __bf16 As[128*LDK];
  __shared__ alignas(16) __bf16 Bs[128*LDK];
  const int lane = tid & 31, wid = tid >> 5;
  const int wi = wid & 3, wj = wid >> 2;
  const int khalf = lane >> 4, l15 = lane & 15;
  f32x8 acc[2][4] = {};
  const float* Ab = prod_s + (size_t)b*NN*NN;
  for (int ks = 0; ks < NN; ks += 32) {
    for (int q = 0; q < 16; ++q) {
      const int idx = q*256 + tid;          // 4096 = 128 rows x 32 k
      const int r = idx >> 5, m = idx & 31;
      float av = Ab[(size_t)(i0 + r)*NN + ks + m] + b_s[(i0 + r)*NN + ks + m];
      As[r*LDK + m] = (__bf16)sigmoidf_(av);
      Bs[r*LDK + m] = (__bf16)V_s[(size_t)(j0 + r)*NN + ks + m];
    }
    __syncthreads();
    bf16x16 af[2], bfr[4];
    for (int mi = 0; mi < 2; ++mi)
      af[mi] = ldsfrag(&As[(wi*32 + mi*16 + l15)*LDK], khalf*8, 16 + khalf*8);
    for (int nj = 0; nj < 4; ++nj)
      bfr[nj] = ldsfrag(&Bs[(wj*64 + nj*16 + l15)*LDK], khalf*8, 16 + khalf*8);
    for (int mi = 0; mi < 2; ++mi)
      for (int nj = 0; nj < 4; ++nj)
        acc[mi][nj] = __builtin_amdgcn_wmma_f32_16x16x32_bf16(
            false, af[mi], false, bfr[nj], (short)0, acc[mi][nj], false, false);
    __syncthreads();
  }
  for (int mi = 0; mi < 2; ++mi)
    for (int nj = 0; nj < 4; ++nj) {
      const int ig = i0 + wi*32 + mi*16 + khalf*8;
      const int jg = j0 + wj*64 + nj*16 + l15;
      for (int r = 0; r < 8; ++r)
        S[((size_t)b*NN + ig + r)*NN + jg] = acc[mi][nj][r];
    }
}

// ---------------------------------------------------------------------------
// k5b: softmax over axis 1 (rows i), one thread per column j (coalesced lanes).
// ---------------------------------------------------------------------------
__global__ void k5b_softmax(float* __restrict__ S) {
  const int b = blockIdx.y;
  const int j = blockIdx.x*256 + threadIdx.x;
  float* col = S + (size_t)b*NN*NN + j;
  float m = -1e30f;
  for (int i = 0; i < NN; ++i) m = fmaxf(m, col[(size_t)i*NN]);
  float s = 0.f;
  for (int i = 0; i < NN; ++i) s += __expf(col[(size_t)i*NN] - m);
  const float inv = 1.f/s;
  for (int i = 0; i < NN; ++i) col[(size_t)i*NN] = __expf(col[(size_t)i*NN] - m) * inv;
}

// ---------------------------------------------------------------------------
// k6: G[b,i,ct] = sum_m (cheb_k[m,i]*S[b,m,i]) * x[b,m,ct]  (A^T GEMM, K=1024)
// WMMA bf16; elementwise cheb*S fused into A-tile load with LDS transpose.
// ---------------------------------------------------------------------------
__global__ void k6_chebG(const float* __restrict__ cheb, const float* __restrict__ S,
                         const float* __restrict__ x, float* __restrict__ G, int kk) {
  const int b = blockIdx.z, i0 = blockIdx.y*128, ct0 = blockIdx.x*128, tid = threadIdx.x;
  __shared__ alignas(16) __bf16 As[128*LDK];
  __shared__ alignas(16) __bf16 Bs[128*LDK];
  const int lane = tid & 31, wid = tid >> 5;
  const int wi = wid & 3, wj = wid >> 2;
  const int khalf = lane >> 4, l15 = lane & 15;
  f32x8 acc[2][4] = {};
  const float* chk = cheb + (size_t)kk*NN*NN;
  const float* Sb  = S + (size_t)b*NN*NN;
  const float* xb  = x + (size_t)b*NN*CT;
  for (int ks = 0; ks < NN; ks += 32) {
    if (ks + 32 < NN)  // hint the next X tile toward the caches
      __builtin_prefetch(xb + (size_t)(ks + 32)*CT + ct0, 0, 1);
    for (int q = 0; q < 16; ++q) {
      const int idx = q*256 + tid;          // 4096 = 32 m-rows x 128 cols
      const int m = idx >> 7, e = idx & 127;
      As[e*LDK + m] = (__bf16)(chk[(size_t)(ks+m)*NN + i0 + e] * Sb[(size_t)(ks+m)*NN + i0 + e]);
      Bs[e*LDK + m] = (__bf16)xb[(size_t)(ks+m)*CT + ct0 + e];
    }
    __syncthreads();
    bf16x16 af[2], bfr[4];
    for (int mi = 0; mi < 2; ++mi)
      af[mi] = ldsfrag(&As[(wi*32 + mi*16 + l15)*LDK], khalf*8, 16 + khalf*8);
    for (int nj = 0; nj < 4; ++nj)
      bfr[nj] = ldsfrag(&Bs[(wj*64 + nj*16 + l15)*LDK], khalf*8, 16 + khalf*8);
    for (int mi = 0; mi < 2; ++mi)
      for (int nj = 0; nj < 4; ++nj)
        acc[mi][nj] = __builtin_amdgcn_wmma_f32_16x16x32_bf16(
            false, af[mi], false, bfr[nj], (short)0, acc[mi][nj], false, false);
    __syncthreads();
  }
  for (int mi = 0; mi < 2; ++mi)
    for (int nj = 0; nj < 4; ++nj) {
      const int ig = i0 + wi*32 + mi*16 + khalf*8;
      const int cg = ct0 + wj*64 + nj*16 + l15;
      for (int r = 0; r < 8; ++r)
        G[((size_t)b*NN + ig + r)*CT + cg] = acc[mi][nj][r];
    }
}

// ---------------------------------------------------------------------------
// k7: accum[b,i,f,t] (+)= sum_c G[b,i,c,t] * Theta[k][c,f]
// ---------------------------------------------------------------------------
__global__ void k7_theta(const float* __restrict__ G, const float* __restrict__ Theta,
                         float* __restrict__ accum, int kk, int beta) {
  const int b = blockIdx.y, i0 = blockIdx.x*4, tid = threadIdx.x;
  __shared__ float Gt[4*CT];
  __shared__ float Th[CC*FC];
  const float* Gb = G + ((size_t)b*NN + i0)*CT;
  for (int idx = tid; idx < 4*CT; idx += 256) Gt[idx] = Gb[idx];
  for (int idx = tid; idx < CC*FC; idx += 256) Th[idx] = Theta[kk*CC*FC + idx];
  __syncthreads();
  float* ob = accum + ((size_t)b*NN + i0)*CT;
  for (int q = 0; q < 24; ++q) {
    const int o = q*256 + tid;
    const int il = o / CT, rem = o % CT;
    const int f = rem / TT, t = rem % TT;
    float v = 0.f;
    for (int c = 0; c < CC; ++c) v += Th[c*FC + f] * Gt[il*CT + c*TT + t];
    if (beta) ob[o] += v; else ob[o] = v;
  }
}

// ---------------------------------------------------------------------------
// k8: spatial_gcn = relu(accum); tco = 1x3 time conv; rco = 1x1 residual on x;
//     v = relu(tco+rco); LayerNorm over Ft. Grid (N/4, B), 4 n per block.
// ---------------------------------------------------------------------------
__global__ void k8_final(const float* __restrict__ accum, const float* __restrict__ x,
                         const float* __restrict__ tc_w, const float* __restrict__ tc_b,
                         const float* __restrict__ rc_w, const float* __restrict__ rc_b,
                         const float* __restrict__ ln_g, const float* __restrict__ ln_b,
                         float* __restrict__ out) {
  const int b = blockIdx.y, n0 = blockIdx.x*4, tid = threadIdx.x;
  __shared__ float sg[4*CT];
  __shared__ float xr[4*CT];
  __shared__ float mu_s[4*TT], rs_s[4*TT];
  const size_t base = ((size_t)b*NN + n0)*CT;
  for (int idx = tid; idx < 4*CT; idx += 256) {
    sg[idx] = fmaxf(accum[base + idx], 0.f);
    xr[idx] = x[base + idx];
  }
  __syncthreads();
  float vbuf[24];
  for (int q = 0; q < 24; ++q) {
    const int o = q*256 + tid;
    const int il = o / CT, rem = o % CT;
    const int f = rem / TT, t = rem % TT;
    float tco = tc_b[f];
    const float* wrow = tc_w + f*FC*3;
    const float* srow = sg + il*CT;
    for (int fc = 0; fc < FC; ++fc) {
      const float s0 = (t > 0)      ? srow[fc*TT + t - 1] : 0.f;
      const float s1 = srow[fc*TT + t];
      const float s2 = (t < TT - 1) ? srow[fc*TT + t + 1] : 0.f;
      tco += s0*wrow[fc*3+0] + s1*wrow[fc*3+1] + s2*wrow[fc*3+2];
    }
    float rco = rc_b[f];
    const float* xrow = xr + il*CT;
    const float* rw = rc_w + f*CC;
    for (int c = 0; c < CC; ++c) rco += xrow[c*TT + t] * rw[c];
    vbuf[q] = fmaxf(tco + rco, 0.f);
  }
  __syncthreads();
  for (int q = 0; q < 24; ++q) sg[q*256 + tid] = vbuf[q];
  __syncthreads();
  if (tid < 96) {
    const int il = tid / TT, t = tid % TT;
    float m = 0.f, m2 = 0.f;
    for (int f = 0; f < FC; ++f) { const float v = sg[il*CT + f*TT + t]; m += v; m2 += v*v; }
    m *= (1.f/FC); m2 *= (1.f/FC);
    mu_s[tid] = m;
    rs_s[tid] = rsqrtf(fmaxf(m2 - m*m, 0.f) + 1e-5f);
  }
  __syncthreads();
  for (int q = 0; q < 24; ++q) {
    const int o = q*256 + tid;
    const int il = o / CT, rem = o % CT;
    const int f = rem / TT, t = rem % TT;
    out[base + o] = (sg[o] - mu_s[il*TT + t]) * rs_s[il*TT + t] * ln_g[f] + ln_b[f];
  }
}

extern "C" void kernel_launch(void* const* d_in, const int* in_sizes, int n_in,
                              void* d_out, int out_size, void* d_ws, size_t ws_size,
                              hipStream_t stream) {
  (void)in_sizes; (void)n_in; (void)out_size; (void)ws_size;
  const float* x     = (const float*)d_in[0];
  const float* cheb  = (const float*)d_in[1];
  const float* U1    = (const float*)d_in[2];
  const float* U2    = (const float*)d_in[3];
  const float* U3    = (const float*)d_in[4];
  const float* b_e   = (const float*)d_in[5];
  const float* V_e   = (const float*)d_in[6];
  const float* W1    = (const float*)d_in[7];
  const float* W2    = (const float*)d_in[8];
  const float* W3    = (const float*)d_in[9];
  const float* b_s   = (const float*)d_in[10];
  const float* V_s   = (const float*)d_in[11];
  const float* Theta = (const float*)d_in[12];
  const float* tc_w  = (const float*)d_in[13];
  const float* tc_b  = (const float*)d_in[14];
  const float* rc_w  = (const float*)d_in[15];
  const float* rc_b  = (const float*)d_in[16];
  const float* ln_g  = (const float*)d_in[17];
  const float* ln_b  = (const float*)d_in[18];
  float* out = (float*)d_out;
  float* ws  = (float*)d_ws;

  // workspace layout (floats)
  float* E    = ws;                    // 32*576
  float* ew   = E    + 18432;          // 32*24
  float* a_tc = ew   + 768;            // 32*1536
  float* r_ct = a_tc + 49152;          // 32*1536
  float* lhs  = r_ct + 49152;          // 32*1024*24
  float* rhs  = lhs  + 786432;         // 32*24*1024
  float* prod = rhs  + 786432;         // 32*1024*1024
  float* S    = prod + 33554432;       // 32*1024*1024
  float* G    = S    + 33554432;       // 32*1024*1536
  float* acc  = G    + 50331648;       // 32*1024*1536

  hipMemsetAsync(a_tc, 0, (size_t)2*49152*sizeof(float), stream);
  k1_temporal_pre<<<dim3(16, BB), 256, 0, stream>>>(x, U1, U2, U3, a_tc, r_ct);
  k2_E<<<BB, 576, 0, stream>>>(a_tc, r_ct, b_e, V_e, W1, E, ew);
  k3_spatial_pre<<<dim3(NN/8, BB), 256, 0, stream>>>(x, E, ew, W2, W3, lhs, rhs);
  k4_prods<<<dim3(16, 16, BB), 256, 0, stream>>>(lhs, rhs, prod);
  k5a_spre<<<dim3(8, 8, BB), 256, 0, stream>>>(prod, b_s, V_s, S);
  k5b_softmax<<<dim3(4, BB), 256, 0, stream>>>(S);
  for (int kk = 0; kk < 3; ++kk) {
    k6_chebG<<<dim3(12, 8, BB), 256, 0, stream>>>(cheb, S, x, G, kk);
    k7_theta<<<dim3(NN/4, BB), 256, 0, stream>>>(G, Theta, acc, kk, kk > 0 ? 1 : 0);
  }
  k8_final<<<dim3(NN/4, BB), 256, 0, stream>>>(acc, x, tc_w, tc_b, rc_w, rc_b, ln_g, ln_b, out);
}
